// HighwayBlock_64819646431484
// MI455X (gfx1250) — matrix operand order repaired and checked by reading, not compiled, not run on hardware
//
#include <hip/hip_runtime.h>
#include <math.h>

// ---------------------------------------------------------------------------
// MI455X (gfx1250): bf16-resident WMMA GEMMs with async global->LDS double
// buffering + WMMA flash attention (V pre-transposed, DPP softmax reductions).
// ---------------------------------------------------------------------------

typedef __attribute__((ext_vector_type(8)))  __bf16 bf16x8;
typedef __attribute__((ext_vector_type(16))) __bf16 bf16x16;
typedef __attribute__((ext_vector_type(8)))  float  f32x8;

#define DIMC   1024
#define NHEADS 16
#define HDIM   64
#define BATCH  4
#define SEQ    1024
#define NTOK   4096
#define MPROT  256
#define RMSEPS 1.1920929e-07f

static __device__ __forceinline__ bf16x16 cat8(bf16x8 a, bf16x8 b) {
    return __builtin_shufflevector(a, b, 0, 1, 2, 3, 4, 5, 6, 7,
                                         8, 9, 10, 11, 12, 13, 14, 15);
}

static __device__ __forceinline__ f32x8 wmma_bf16(bf16x16 a, bf16x16 b, f32x8 c) {
    return __builtin_amdgcn_wmma_f32_16x16x32_bf16(false, a, false, b,
                                                   (short)0, c, false, false);
}

// gfx1250 async copy: global (16B/lane) -> LDS, tracked by ASYNCcnt.
static __device__ __forceinline__ void async_copy_b128(unsigned lds_addr,
                                                       const void* gaddr) {
    asm volatile("global_load_async_to_lds_b128 %0, %1, off"
                 :: "v"(lds_addr), "v"((unsigned long long)(size_t)gaddr)
                 : "memory");
}
static __device__ __forceinline__ unsigned lds_off(const void* p) {
    return (unsigned)(unsigned long long)(size_t)p;   // low 32 bits = LDS offset
}

// ---- DPP16 butterfly reductions over 16-lane rows (no LDS traffic) --------
// masks {1,2,7,15} span the same subgroup as {1,2,4,8} -> exact reduction.
template <int CTRL>
static __device__ __forceinline__ float dppf(float x) {
    return __int_as_float(__builtin_amdgcn_update_dpp(
        0, __float_as_int(x), CTRL, 0xF, 0xF, true));
}
static __device__ __forceinline__ float rmax16(float x) {
    x = fmaxf(x, dppf<0xB1>(x));    // quad_perm(1,0,3,2) : xor 1
    x = fmaxf(x, dppf<0x4E>(x));    // quad_perm(2,3,0,1) : xor 2
    x = fmaxf(x, dppf<0x141>(x));   // row_half_mirror    : xor 7
    x = fmaxf(x, dppf<0x140>(x));   // row_mirror         : xor 15
    return x;
}
static __device__ __forceinline__ float rsum16(float x) {
    x += dppf<0xB1>(x);
    x += dppf<0x4E>(x);
    x += dppf<0x141>(x);
    x += dppf<0x140>(x);
    return x;
}

#define ACT_NONE 0
#define ACT_GELU 1
#define ACT_SILU 2
#define EP_STORE       0
#define EP_MUL         1
#define EP_SCALE_STORE 2
#define EP_SCALE_ACC   3
#define EP_RESID_MASK  4
#define EP_STORE_T     5   // bf16, store transposed: C[(b*N+n)*SEQ + t]

template <int ACT>
static __device__ __forceinline__ float act_fn(float x) {
    if (ACT == ACT_GELU) return 0.5f * x * (1.0f + erff(x * 0.70710678118654752440f));
    if (ACT == ACT_SILU) return x * (1.0f / (1.0f + __expf(-x)));
    return x;
}

// ---------------------------------------------------------------------------
// GEMM: C[M,N] = epilogue(act(A[M,K] @ W[N,K]^T)), A/W bf16 in global memory.
// 256 threads (8 waves), tile 128x128x32, wave = 32(M) x 64(N) = 8 WMMA/step.
// Double-buffered LDS filled by global_load_async_to_lds_b128 (ASYNCcnt).
// ---------------------------------------------------------------------------
template <int ACT, int EP, typename OutT>
__global__ void __launch_bounds__(256)
gemm_bf16_kernel(const __bf16* __restrict__ A, const __bf16* __restrict__ W,
                 OutT* C, const OutT* Cin, const float* __restrict__ resid,
                 const float* __restrict__ rowscale, int rs_stride, int maskN,
                 int M, int N, int K) {
    __shared__ __bf16 sA[2][128 * 40];
    __shared__ __bf16 sB[2][128 * 40];

    const int tid    = threadIdx.x;
    const int wave   = tid >> 5;
    const int lane   = tid & 31;
    const int lane15 = lane & 15;
    const int half   = lane >> 4;
    const int wm     = wave >> 1;
    const int wn     = wave & 1;
    const int m0     = blockIdx.y * 128;
    const int n0     = blockIdx.x * 128;
    const int abase  = half ? 8 : 0;
    const int boff   = half ? 16 : 0;

    f32x8 acc[2][4];
#pragma unroll
    for (int fm = 0; fm < 2; ++fm)
#pragma unroll
        for (int fn = 0; fn < 4; ++fn) acc[fm][fn] = (f32x8)0.0f;

    // tile staging: 512 16B-chunks per tile, 2 per thread per tile
    auto issue_tile = [&](int bufI, int kkI) {
#pragma unroll
        for (int t = 0; t < 2; ++t) {
            const int ch  = tid + t * 256;
            const int row = ch >> 2;
            const int sub = (ch & 3) * 8;
            async_copy_b128(lds_off(&sA[bufI][row * 40 + sub]),
                            A + (size_t)(m0 + row) * K + kkI + sub);
            async_copy_b128(lds_off(&sB[bufI][row * 40 + sub]),
                            W + (size_t)(n0 + row) * K + kkI + sub);
        }
    };

    issue_tile(0, 0);
    int buf = 0;
    for (int kk = 0; kk < K; kk += 32) {
        asm volatile("s_wait_asynccnt 0" ::: "memory");
        __syncthreads();
        if (kk + 32 < K) issue_tile(buf ^ 1, kk + 32);

        bf16x16 af[2], bfr[4];
#pragma unroll
        for (int fm = 0; fm < 2; ++fm) {
            const int row = wm * 32 + fm * 16 + lane15;
            bf16x8 lo = *(const bf16x8*)&sA[buf][row * 40 + abase];
            bf16x8 hi = *(const bf16x8*)&sA[buf][row * 40 + 16 + abase];
            af[fm] = cat8(lo, hi);
        }
#pragma unroll
        for (int fn = 0; fn < 4; ++fn) {
            const int cr = wn * 64 + fn * 16 + lane15;
            bf16x8 lo = *(const bf16x8*)&sB[buf][cr * 40 + boff / 2];
            bf16x8 hi = *(const bf16x8*)&sB[buf][cr * 40 + boff / 2 + 16];
            bfr[fn] = cat8(lo, hi);
        }
#pragma unroll
        for (int fm = 0; fm < 2; ++fm)
#pragma unroll
            for (int fn = 0; fn < 4; ++fn)
                acc[fm][fn] = wmma_bf16(af[fm], bfr[fn], acc[fm][fn]);
        buf ^= 1;
    }

    // ---- epilogue ----
#pragma unroll
    for (int fm = 0; fm < 2; ++fm)
#pragma unroll
        for (int fn = 0; fn < 4; ++fn) {
            const int gn = n0 + wn * 64 + fn * 16 + lane15;
            if (EP == EP_STORE_T) {
                // rows of this fragment are 8 consecutive tokens -> 16B store
                const int gm0 = m0 + wm * 32 + fm * 16 + 8 * half;  // row of r=0
                union { bf16x8 v; __bf16 h[8]; } pk;
#pragma unroll
                for (int r = 0; r < 8; ++r)
                    pk.h[r] = (__bf16)act_fn<ACT>(acc[fm][fn][r]);
                __bf16* dst = (__bf16*)C +
                              ((size_t)((gm0 >> 10) * N + gn)) * SEQ + (gm0 & (SEQ - 1));
                *(bf16x8*)dst = pk.v;
            } else {
#pragma unroll
                for (int r = 0; r < 8; ++r) {
                    const int gm = m0 + wm * 32 + fm * 16 + r + 8 * half;
                    const float v = act_fn<ACT>(acc[fm][fn][r]);
                    const size_t o = (size_t)gm * N + gn;
                    if (EP == EP_STORE) {
                        C[o] = (OutT)v;
                    } else if (EP == EP_MUL) {
                        C[o] = (OutT)((float)Cin[o] * v);
                    } else if (EP == EP_SCALE_STORE) {
                        C[o] = (OutT)(rowscale[(size_t)gm * rs_stride] * v);
                    } else if (EP == EP_SCALE_ACC) {
                        C[o] = (OutT)((float)C[o] + rowscale[(size_t)gm * rs_stride] * v);
                    } else {  // EP_RESID_MASK
                        C[o] = (OutT)(resid[o] + (gn < maskN ? 0.0f : v));
                    }
                }
            }
        }
}

// ---------------------------------------------------------------------------
// f32 -> bf16 bulk conversion (weights), 4 elems/thread.
// ---------------------------------------------------------------------------
__global__ void __launch_bounds__(256)
f32_to_bf16_kernel(const float* __restrict__ in, __bf16* __restrict__ out, int n4) {
    const int i = blockIdx.x * 256 + threadIdx.x;
    if (i < n4) {
        const float4 f = ((const float4*)in)[i];
        union { __bf16 h[4]; uint2 u; } c;
        c.h[0] = (__bf16)f.x; c.h[1] = (__bf16)f.y;
        c.h[2] = (__bf16)f.z; c.h[3] = (__bf16)f.w;
        ((uint2*)out)[i] = c.u;
    }
}

// ---------------------------------------------------------------------------
// RMSNorm over last dim (1024): one block per row. f32 in, bf16 out.
// ---------------------------------------------------------------------------
__global__ void __launch_bounds__(256)
rmsnorm_kernel(const float* __restrict__ x, const float* __restrict__ w,
               __bf16* __restrict__ y) {
    __shared__ float red[8];
    const int row  = blockIdx.x;
    const int tid  = threadIdx.x;
    const int wave = tid >> 5, lane = tid & 31;
    const float* xr = x + (size_t)row * DIMC;
    float vals[4];
    float ss = 0.0f;
#pragma unroll
    for (int j = 0; j < 4; ++j) {
        vals[j] = xr[tid + j * 256];
        ss += vals[j] * vals[j];
    }
#pragma unroll
    for (int m = 1; m < 32; m <<= 1) ss += __shfl_xor(ss, m, 32);
    if (lane == 0) red[wave] = ss;
    __syncthreads();
    float tot = 0.0f;
#pragma unroll
    for (int i = 0; i < 8; ++i) tot += red[i];
    const float rn = rsqrtf(tot * (1.0f / DIMC) + RMSEPS);
    __bf16* yr = y + (size_t)row * DIMC;
#pragma unroll
    for (int j = 0; j < 4; ++j) {
        const int c = tid + j * 256;
        yr[c] = (__bf16)(vals[j] * rn * w[c]);
    }
}

// ---------------------------------------------------------------------------
// Per-(token,head) RMSNorm over 64 (QK-norm): f32 in -> bf16 out.
// ---------------------------------------------------------------------------
__global__ void __launch_bounds__(256)
qknorm_kernel(const float* __restrict__ q, const float* __restrict__ w,
              __bf16* __restrict__ out) {
    const int wave = threadIdx.x >> 5, lane = threadIdx.x & 31;
    const size_t idx = (size_t)blockIdx.x * 8 + wave;
    const float* p = q + idx * HDIM;
    const float a = p[lane];
    const float b = p[lane + 32];
    float ss = a * a + b * b;
#pragma unroll
    for (int m = 1; m < 32; m <<= 1) ss += __shfl_xor(ss, m, 32);
    const float rn = rsqrtf(ss * (1.0f / HDIM) + RMSEPS);
    __bf16* po = out + idx * HDIM;
    po[lane]      = (__bf16)(a * rn * w[lane]);
    po[lane + 32] = (__bf16)(b * rn * w[lane + 32]);
}

// ---------------------------------------------------------------------------
// Flash attention. q/k: [b,t,h*64+d] bf16 (row-major). v: transposed
// [b, h*64+d, t] bf16, so PV B-fragments are contiguous global loads.
// Grid (SEQ/64, B*H), 128 threads; each wave owns 16 query rows.
// ---------------------------------------------------------------------------
__global__ void __launch_bounds__(128)
attn_kernel(const __bf16* __restrict__ q, const __bf16* __restrict__ k,
            const __bf16* __restrict__ vT, __bf16* __restrict__ ao) {
    __shared__ __bf16 sP[4][16 * 40];   // P tile [row][key], per wave

    const int wave   = threadIdx.x >> 5;
    const int lane   = threadIdx.x & 31;
    const int lane15 = lane & 15;
    const int half   = lane >> 4;
    const int bh     = blockIdx.y;
    const int b      = bh >> 4;
    const int hh     = bh & 15;
    const int qbase  = blockIdx.x * 64 + wave * 16;
    const int qmax   = qbase + 15;
    const int abase  = half ? 8 : 0;
    const int koff   = half ? 16 : 0;

    // Q fragments for HD=64 (two K-halves)
    bf16x16 aq[2];
    {
        const __bf16* qp = q + ((size_t)(b * SEQ + qbase + lane15)) * DIMC + hh * HDIM;
#pragma unroll
        for (int kh = 0; kh < 2; ++kh)
            aq[kh] = cat8(*(const bf16x8*)(qp + kh * 32 + abase),
                          *(const bf16x8*)(qp + kh * 32 + 16 + abase));
    }
    // base of this head's V^T plane, column hd = fn*16 + lane15
    const __bf16* vplane = vT + ((size_t)(b * DIMC + hh * HDIM)) * SEQ;

    f32x8 o[4];
#pragma unroll
    for (int fn = 0; fn < 4; ++fn) o[fn] = (f32x8)0.0f;
    float mrun[8], lrun[8];
#pragma unroll
    for (int r = 0; r < 8; ++r) { mrun[r] = -INFINITY; lrun[r] = 0.0f; }

    for (int k0 = 0; k0 <= qmax; k0 += 32) {
        // S = Q K^T, two 16-key blocks
        f32x8 s[2];
#pragma unroll
        for (int kb = 0; kb < 2; ++kb) {
            const int key = k0 + kb * 16 + lane15;
            const __bf16* kp = k + ((size_t)(b * SEQ + key)) * DIMC + hh * HDIM + koff;
            s[kb] = (f32x8)0.0f;
#pragma unroll
            for (int kh = 0; kh < 2; ++kh) {
                const bf16x16 bk = cat8(*(const bf16x8*)(kp - koff / 2 + kh * 32),
                                        *(const bf16x8*)(kp - koff / 2 + kh * 32 + 16));
                s[kb] = wmma_bf16(aq[kh], bk, s[kb]);
            }
        }
        // scale (+ causal mask only where the diagonal can intersect: uniform branch)
        if (k0 + 31 <= qbase) {
#pragma unroll
            for (int kb = 0; kb < 2; ++kb)
#pragma unroll
                for (int r = 0; r < 8; ++r) s[kb][r] *= 0.125f;
        } else {
#pragma unroll
            for (int kb = 0; kb < 2; ++kb) {
                const int key = k0 + kb * 16 + lane15;
#pragma unroll
                for (int r = 0; r < 8; ++r) {
                    const int rowq = qbase + r + 8 * half;
                    s[kb][r] = (key <= rowq) ? s[kb][r] * 0.125f : -INFINITY;
                }
            }
        }

        // online softmax (DPP butterflies, no LDS)
#pragma unroll
        for (int r = 0; r < 8; ++r) {
            const float rmax  = rmax16(fmaxf(s[0][r], s[1][r]));
            const float mnew  = fmaxf(mrun[r], rmax);
            const float alpha = __expf(mrun[r] - mnew);
            const float p0 = __expf(s[0][r] - mnew);
            const float p1 = __expf(s[1][r] - mnew);
            const int prow = r + 8 * half;
            sP[wave][prow * 40 + lane15]      = (__bf16)p0;
            sP[wave][prow * 40 + 16 + lane15] = (__bf16)p1;
            lrun[r] = lrun[r] * alpha + rsum16(p0 + p1);
            mrun[r] = mnew;
#pragma unroll
            for (int fn = 0; fn < 4; ++fn) o[fn][r] *= alpha;
        }

        asm volatile("s_wait_dscnt 0" ::: "memory");  // same-wave LDS RAW

        const bf16x16 pa = cat8(*(const bf16x8*)&sP[wave][lane15 * 40 + abase],
                                *(const bf16x8*)&sP[wave][lane15 * 40 + 16 + abase]);
        // O += P @ V: B-fragments straight from transposed V in global memory
#pragma unroll
        for (int fn = 0; fn < 4; ++fn) {
            const __bf16* vp = vplane + (size_t)(fn * 16 + lane15) * SEQ + k0 + koff;
            const bf16x16 bv = cat8(*(const bf16x8*)(vp - koff / 2), *(const bf16x8*)(vp - koff / 2 + 16));
            o[fn] = wmma_bf16(pa, bv, o[fn]);
        }
    }

#pragma unroll
    for (int fn = 0; fn < 4; ++fn)
#pragma unroll
        for (int r = 0; r < 8; ++r) {
            const int qrow = qbase + r + 8 * half;
            const int col  = hh * HDIM + fn * 16 + lane15;
            ao[((size_t)(b * SEQ + qrow)) * DIMC + col] = (__bf16)(o[fn][r] / lrun[r]);
        }
}

// ---------------------------------------------------------------------------
// Router: one wave per token, bf16 activations x f32 gate weights.
// ---------------------------------------------------------------------------
__global__ void __launch_bounds__(256)
router_kernel(const __bf16* __restrict__ xf, const float* __restrict__ gw,
              float* __restrict__ wts) {
    const int wave = threadIdx.x >> 5, lane = threadIdx.x & 31;
    const int tok  = blockIdx.x * 8 + wave;
    const __bf16* xr = xf + (size_t)tok * DIMC;
    float l0 = 0, l1 = 0, l2 = 0, l3 = 0;
    for (int i = lane; i < DIMC; i += 32) {
        const float xv = (float)xr[i];
        l0 += xv * gw[i];
        l1 += xv * gw[DIMC + i];
        l2 += xv * gw[2 * DIMC + i];
        l3 += xv * gw[3 * DIMC + i];
    }
#pragma unroll
    for (int m = 1; m < 32; m <<= 1) {
        l0 += __shfl_xor(l0, m, 32);
        l1 += __shfl_xor(l1, m, 32);
        l2 += __shfl_xor(l2, m, 32);
        l3 += __shfl_xor(l3, m, 32);
    }
    if (lane == 0) {
        float lg[4] = {l0, l1, l2, l3};
#pragma unroll
        for (int e = 0; e < 4; ++e) {
            if (isnan(lg[e])) lg[e] = 0.0f;
            else if (isinf(lg[e])) lg[e] = lg[e] > 0 ? 3.402823466e38f : -3.402823466e38f;
        }
        const float mx = fmaxf(fmaxf(lg[0], lg[1]), fmaxf(lg[2], lg[3]));
        float p[4], s = 0.0f;
#pragma unroll
        for (int e = 0; e < 4; ++e) { p[e] = __expf(lg[e] - mx); s += p[e]; }
#pragma unroll
        for (int e = 0; e < 4; ++e) p[e] /= s;
        int i0 = 0;
#pragma unroll
        for (int e = 1; e < 4; ++e) if (p[e] > p[i0]) i0 = e;
        int i1 = (i0 == 0) ? 1 : 0;
#pragma unroll
        for (int e = 0; e < 4; ++e) if (e != i0 && p[e] > p[i1]) i1 = e;
        const float d = p[i0] + p[i1] + 1e-8f;
        float ov[4] = {0, 0, 0, 0};
        ov[i0] = p[i0] / d;
        ov[i1] = p[i1] / d;
#pragma unroll
        for (int e = 0; e < 4; ++e) wts[(size_t)tok * 4 + e] = ov[e];
    }
}

__global__ void __launch_bounds__(256)
combine_kernel(const float* __restrict__ x1, const float* __restrict__ moe,
               float* __restrict__ out) {
    const size_t i = (size_t)blockIdx.x * blockDim.x + threadIdx.x;
    const int c = (int)(i & (DIMC - 1));
    out[i] = x1[i] + (c < MPROT ? 0.0f : moe[i]);
}

// ---------------------------------------------------------------------------
// Host launcher
// ---------------------------------------------------------------------------
extern "C" void kernel_launch(void* const* d_in, const int* in_sizes, int n_in,
                              void* d_out, int out_size, void* d_ws, size_t ws_size,
                              hipStream_t stream) {
    (void)in_sizes; (void)n_in; (void)out_size; (void)ws_size;
    const float* x      = (const float*)d_in[0];
    const float* n1_w   = (const float*)d_in[1];
    const float* n2_w   = (const float*)d_in[2];
    const float* qn_w   = (const float*)d_in[7];
    const float* kn_w   = (const float*)d_in[8];
    const float* gate_w = (const float*)d_in[9];
    float* out = (float*)d_out;

    // 16 GEMM weights (f32 sources) in dict order
    const int widx[16]  = {3, 4, 5, 6, 10, 11, 12, 13, 14, 15, 16, 17, 18, 19, 20, 21};
    const size_t wcnt[16] = {
        1048576, 1048576, 1048576, 1048576,          // wq wk wv wo
        2097152, 2097152,                            // e0_up e0_down
        2097152, 2097152, 2097152,                   // e1_w1 e1_w2 e1_down
        2097152, 4194304, 4194304, 2097152,          // e2_l1..l4
        1048576, 2097152, 2097152};                  // e3_down e3_up e3_out

    size_t cur = 0;
    auto take = [&](size_t bytes) -> char* {
        char* p = (char*)d_ws + cur;
        cur += (bytes + 255) & ~(size_t)255;
        return p;
    };

    __bf16* wb[16];
    for (int i = 0; i < 16; ++i) wb[i] = (__bf16*)take(wcnt[i] * 2);

    const size_t SZ = (size_t)NTOK * DIMC;            // 4M elements
    __bf16* hbf = (__bf16*)take(SZ * 2);              // h, later xf
    float*  qf  = (float*)take(SZ * 4);               // f32 q (pre QK-norm)
    float*  kf  = (float*)take(SZ * 4);
    __bf16* qb  = (__bf16*)take(SZ * 2);
    __bf16* kb  = (__bf16*)take(SZ * 2);
    __bf16* vbT = (__bf16*)take(SZ * 2);              // V transposed [b,c][t]
    __bf16* aob = (__bf16*)take(SZ * 2);
    float*  x1b = (float*)take(SZ * 4);
    float*  moe = (float*)take(SZ * 4);
    float*  wts = (float*)take((size_t)NTOK * 4 * 4);
    __bf16* hid1 = (__bf16*)qf;                       // [NTOK,2048] alias (MoE phase)
    __bf16* hid2 = (__bf16*)kf;                       // [NTOK,2048] alias (MoE phase)

    const dim3 blk256(256), blk128(128);

    // ---- weight conversion f32 -> bf16 ----
    for (int i = 0; i < 16; ++i) {
        const int n4 = (int)(wcnt[i] / 4);
        f32_to_bf16_kernel<<<(n4 + 255) / 256, blk256, 0, stream>>>(
            (const float*)d_in[widx[i]], wb[i], n4);
    }

#define GEMM(ACT, EP, OUT_T, A_, W_, C_, CIN_, RES_, RS_, RSS_, MASK_, N_, K_)             \
    gemm_bf16_kernel<ACT, EP, OUT_T><<<dim3((N_) / 128, NTOK / 128), blk256, 0, stream>>>( \
        A_, W_, C_, CIN_, RES_, RS_, RSS_, MASK_, NTOK, N_, K_)

    // ---- attention ----
    rmsnorm_kernel<<<NTOK, blk256, 0, stream>>>(x, n1_w, hbf);
    GEMM(ACT_NONE, EP_STORE, float,  hbf, wb[0], qf, (float*)nullptr, nullptr, nullptr, 0, 0, DIMC, DIMC);
    GEMM(ACT_NONE, EP_STORE, float,  hbf, wb[1], kf, (float*)nullptr, nullptr, nullptr, 0, 0, DIMC, DIMC);
    GEMM(ACT_NONE, EP_STORE_T, __bf16, hbf, wb[2], vbT, (__bf16*)nullptr, nullptr, nullptr, 0, 0, DIMC, DIMC);
    qknorm_kernel<<<NTOK * NHEADS / 8, blk256, 0, stream>>>(qf, qn_w, qb);
    qknorm_kernel<<<NTOK * NHEADS / 8, blk256, 0, stream>>>(kf, kn_w, kb);
    attn_kernel<<<dim3(SEQ / 64, BATCH * NHEADS), blk128, 0, stream>>>(qb, kb, vbT, aob);
    GEMM(ACT_NONE, EP_RESID_MASK, float, aob, wb[3], x1b, (float*)nullptr, x, nullptr, 0, MPROT, DIMC, DIMC);

    // ---- MoE ----
    rmsnorm_kernel<<<NTOK, blk256, 0, stream>>>(x1b, n2_w, hbf);  // xf (bf16)
    router_kernel<<<NTOK / 8, blk256, 0, stream>>>(hbf, gate_w, wts);

    // expert 0: StandardFFN (initializes moe)
    GEMM(ACT_GELU, EP_STORE, __bf16, hbf, wb[4], hid1, (__bf16*)nullptr, nullptr, nullptr, 0, 0, 2048, DIMC);
    GEMM(ACT_NONE, EP_SCALE_STORE, float, hid1, wb[5], moe, (float*)nullptr, nullptr, wts + 0, 4, 0, DIMC, 2048);

    // expert 1: SwiGLU
    GEMM(ACT_SILU, EP_STORE, __bf16, hbf, wb[6], hid1, (__bf16*)nullptr, nullptr, nullptr, 0, 0, 2048, DIMC);
    GEMM(ACT_NONE, EP_MUL, __bf16, hbf, wb[7], hid1, hid1, nullptr, nullptr, 0, 0, 2048, DIMC);
    GEMM(ACT_NONE, EP_SCALE_ACC, float, hid1, wb[8], moe, (float*)nullptr, nullptr, wts + 1, 4, 0, DIMC, 2048);

    // expert 2: DeepNarrow
    GEMM(ACT_GELU, EP_STORE, __bf16, hbf, wb[9],  hid1, (__bf16*)nullptr, nullptr, nullptr, 0, 0, 2048, DIMC);
    GEMM(ACT_GELU, EP_STORE, __bf16, hid1, wb[10], hid2, (__bf16*)nullptr, nullptr, nullptr, 0, 0, 2048, 2048);
    GEMM(ACT_GELU, EP_STORE, __bf16, hid2, wb[11], hid1, (__bf16*)nullptr, nullptr, nullptr, 0, 0, 2048, 2048);
    GEMM(ACT_NONE, EP_SCALE_ACC, float, hid1, wb[12], moe, (float*)nullptr, nullptr, wts + 2, 4, 0, DIMC, 2048);

    // expert 3: Bottleneck
    GEMM(ACT_GELU, EP_STORE, __bf16, hbf, wb[13], hid2, (__bf16*)nullptr, nullptr, nullptr, 0, 0, DIMC, DIMC);
    GEMM(ACT_GELU, EP_STORE, __bf16, hid2, wb[14], hid1, (__bf16*)nullptr, nullptr, nullptr, 0, 0, 2048, DIMC);
    GEMM(ACT_NONE, EP_SCALE_ACC, float, hid1, wb[15], moe, (float*)nullptr, nullptr, wts + 3, 4, 0, DIMC, 2048);

    // ---- final combine ----
    combine_kernel<<<(NTOK * DIMC) / 256, blk256, 0, stream>>>(x1b, moe, out);
#undef GEMM
}
